// HexagramMoE_46832323395757
// MI455X (gfx1250) — matrix operand order repaired and checked by reading, not compiled, not run on hardware
//
#include <hip/hip_runtime.h>
#include <hip/hip_bf16.h>
#include <math.h>

// HexagramMoE for MI455X (gfx1250, wave32, WMMA), weight-stationary two-pass.
//
// Roofline: 34.4 GFLOP routed compute; 536 MB fp32 expert weights is the
// mandatory HBM stream (23 us @ 23.3 TB/s). A (expert x token-tile) blocking
// re-reads each expert's weights once per tile (~4.3 GB). Instead:
//   pass 1: block = (expert, 128-row W_in slab): slab -> LDS bf16 ONCE,
//           sweep all token tiles; H = silu(X Wi^T + bi) -> global bf16.
//   pass 2: block = (expert, 64-row W_out slab): slab -> LDS bf16 ONCE,
//           sweep token tiles reading H (L2-resident), gated scatter.
// => every weight byte crosses HBM exactly once (~0.65 GB total traffic).

#define NE      64
#define TK      4
#define DM      512
#define DF      2048
#define FFB     128            // W_in rows staged per pass-1 block
#define DB      64             // W_out rows staged per pass-2 block
#define PAD     8              // LDS row padding (elements) -> conflict-free b128
#define LDS1_B  ((size_t)FFB * (DM + PAD) * 2)   // 133120 B
#define LDS2_B  ((size_t)DB  * (DF + PAD) * 2)   // 263168 B (<= 320KB WGP LDS)

typedef __attribute__((ext_vector_type(16))) __bf16 v16bf;
typedef __attribute__((ext_vector_type(8)))  __bf16 v8bf;
typedef __attribute__((ext_vector_type(4)))  __bf16 v4bf;
typedef __attribute__((ext_vector_type(8)))  float  v8f;
typedef __attribute__((ext_vector_type(4)))  float  v4f;   // native clang vector
                                                           // (HIP float4 is a class,
                                                           //  rejected by NT builtin)

// ---- fragment helpers -------------------------------------------------------
__device__ __forceinline__ v16bf combine16(v8bf lo, v8bf hh) {
  v16bf r;
#pragma unroll
  for (int i = 0; i < 8; ++i) { r[i] = lo[i]; r[i + 8] = hh[i]; }
  return r;
}

// A 16x32 bf16 (ISA 7.12.2): lane holds row M=lane&15; lanes<16: K=k0+0..7 and
// k0+16..23; lanes>=16: +8. Two 16B loads per fragment (LDS or global).
__device__ __forceinline__ v16bf load_a_frag(const __bf16* rowp, int k0, int hi) {
  const __bf16* p = rowp + k0 + hi * 8;
  return combine16(*(const v8bf*)p, *(const v8bf*)(p + 16));
}

// B 32x16 bf16: lane holds column N=lane&15; same K order as A: K=k0+0..7 and
// k0+16..23 (+8 for upper lanes). Two 16B LDS loads.
__device__ __forceinline__ v16bf load_b_frag(const __bf16* p) {
  return combine16(*(const v8bf*)p, *(const v8bf*)(p + 16));
}

// fp32 weight slab -> bf16 LDS (row-padded). Non-temporal: stream-once data,
// keep X/H resident in L2 instead.
__device__ __forceinline__ void stage_w(const float* __restrict__ Wg,
                                        __bf16* Wb, int elems,
                                        int colShift, int colMask, int stridePad) {
  for (int i = threadIdx.x * 4; i < elems; i += 256 * 4) {
    int row = i >> colShift, col = i & colMask;
    v4f q = __builtin_nontemporal_load((const v4f*)(Wg + i));
    v4bf w;
    w[0] = (__bf16)q[0]; w[1] = (__bf16)q[1]; w[2] = (__bf16)q[2]; w[3] = (__bf16)q[3];
    *(v4bf*)(Wb + row * stridePad + col) = w;
  }
}

// ---- routing ----------------------------------------------------------------
__global__ void hexmoe_zero(int* counts, int* cursors) {
  int i = threadIdx.x;
  if (i < NE) { counts[i] = 0; cursors[i] = 0; }
}

__global__ void hexmoe_route(const float* __restrict__ hw,
                             int* __restrict__ top_idx, float* __restrict__ top_w,
                             int* __restrict__ counts, int N) {
  int t = blockIdx.x * blockDim.x + threadIdx.x;
  if (t >= N) return;
  float s[NE];
#pragma unroll
  for (int j = 0; j < NE; ++j) s[j] = hw[(size_t)t * NE + j];
  int ids[TK]; float ws[TK]; float sum = 0.f;
#pragma unroll
  for (int k = 0; k < TK; ++k) {
    float best = -INFINITY; int bi = 0;
#pragma unroll
    for (int j = 0; j < NE; ++j)
      if (s[j] > best) { best = s[j]; bi = j; }   // '>' => lowest index wins ties
    ids[k] = bi; ws[k] = best; s[bi] = -INFINITY; sum += best;
  }
  float inv = 1.f / (sum + 1e-8f);
#pragma unroll
  for (int k = 0; k < TK; ++k) {
    top_idx[t * TK + k] = ids[k];
    top_w[t * TK + k]   = ws[k] * inv;
    atomicAdd(&counts[ids[k]], 1);
  }
}

__global__ void hexmoe_scan(const int* __restrict__ counts, int* __restrict__ offsets) {
  if (threadIdx.x == 0) {
    int acc = 0;
    for (int e = 0; e < NE; ++e) { offsets[e] = acc; acc += counts[e]; }
    offsets[NE] = acc;
  }
}

__global__ void hexmoe_scatter(const int* __restrict__ top_idx,
                               const int* __restrict__ offsets,
                               int* __restrict__ cursors,
                               int* __restrict__ grouped, int N) {
  int t = blockIdx.x * blockDim.x + threadIdx.x;
  if (t >= N) return;
  for (int k = 0; k < TK; ++k) {
    int e = top_idx[t * TK + k];
    int pos = atomicAdd(&cursors[e], 1);
    grouped[offsets[e] + pos] = (t << 2) | k;     // (token, slot)
  }
}

__global__ void hexmoe_xcast(const float* __restrict__ x, __bf16* __restrict__ Xbf,
                             int total) {
  int i = (blockIdx.x * blockDim.x + threadIdx.x) * 4;
  if (i >= total) return;
  v4f q = *(const v4f*)(x + i);
  v4bf w;
  w[0] = (__bf16)q[0]; w[1] = (__bf16)q[1]; w[2] = (__bf16)q[2]; w[3] = (__bf16)q[3];
  *(v4bf*)(Xbf + i) = w;
}

// ---- pass 1: H = silu(X @ Wi^T + bi), weight-stationary ---------------------
__global__ void __launch_bounds__(256)
moe_gemm1(const __bf16* __restrict__ Xbf,
          const float* __restrict__ W_in, const float* __restrict__ bias_in,
          const int* __restrict__ counts, const int* __restrict__ offsets,
          const int* __restrict__ grouped, __bf16* __restrict__ Hbuf) {
  extern __shared__ __align__(16) __bf16 Wb[];   // FFB x (DM+PAD) bf16
  const int e  = blockIdx.x;
  const int fb = blockIdx.y;
  const int m  = counts[e];
  if (m == 0) return;                             // uniform
  const int base = offsets[e];
  const int f0   = fb * FFB;
  const int SW   = DM + PAD;

  stage_w(W_in + ((size_t)e * DF + f0) * DM, Wb, FFB * DM, 9, DM - 1, SW);
  __syncthreads();

  const int tid  = threadIdx.x;
  const int wave = tid >> 5;
  const int lane = tid & 31;
  const int hi   = lane >> 4;
  const int ln   = lane & 15;
  const int ntiles = (m + 15) >> 4;

  for (int tile = wave; tile < ntiles; tile += 8) {
    const int r0 = tile * 16;
    int ridx = r0 + ln; if (ridx >= m) ridx = m - 1;     // clamp; masked at store
    const __bf16* arow = Xbf + (size_t)(grouped[base + ridx] >> 2) * DM;

    v8f acc[FFB / 16] = {};
#pragma unroll 2
    for (int ks = 0; ks < DM / 32; ++ks) {
      const int k0 = ks * 32;
      v16bf a = load_a_frag(arow, k0, hi);
#pragma unroll
      for (int ct = 0; ct < FFB / 16; ++ct) {
        v16bf b = load_b_frag(Wb + (ct * 16 + ln) * SW + k0 + hi * 8);
        acc[ct] = __builtin_amdgcn_wmma_f32_16x16x32_bf16(false, a, false, b,
                                                          (short)0, acc[ct], false, false);
      }
    }
#pragma unroll
    for (int ct = 0; ct < FFB / 16; ++ct) {
      const int col = f0 + ct * 16 + ln;
      const float bia = bias_in[e * DF + col];
#pragma unroll
      for (int v = 0; v < 8; ++v) {
        int r = r0 + v + hi * 8;
        if (r < m) {
          float c = acc[ct][v] + bia;
          float h = c / (1.f + __expf(-c));       // silu
          Hbuf[(size_t)(base + r) * DF + col] = (__bf16)h;
        }
      }
    }
  }
}

// ---- pass 2: O = H @ Wo^T + bo, gated scatter, weight-stationary ------------
__global__ void __launch_bounds__(256)
moe_gemm2(const __bf16* __restrict__ Hbuf,
          const float* __restrict__ W_out, const float* __restrict__ bias_out,
          const int* __restrict__ counts, const int* __restrict__ offsets,
          const int* __restrict__ grouped, const float* __restrict__ top_w,
          float* __restrict__ contrib) {
  extern __shared__ __align__(16) __bf16 Wb[];   // DB x (DF+PAD) bf16
  const int e  = blockIdx.x;
  const int db = blockIdx.y;
  const int m  = counts[e];
  if (m == 0) return;
  const int base = offsets[e];
  const int d0   = db * DB;
  const int SW   = DF + PAD;

  stage_w(W_out + ((size_t)e * DM + d0) * DF, Wb, DB * DF, 11, DF - 1, SW);
  __syncthreads();

  const int tid  = threadIdx.x;
  const int wave = tid >> 5;
  const int lane = tid & 31;
  const int hi   = lane >> 4;
  const int ln   = lane & 15;
  const int ntiles = (m + 15) >> 4;

  for (int tile = wave; tile < ntiles; tile += 8) {
    const int r0 = tile * 16;
    int ridx = r0 + ln; if (ridx >= m) ridx = m - 1;
    const __bf16* arow = Hbuf + (size_t)(base + ridx) * DF;

    v8f acc[DB / 16] = {};
#pragma unroll 2
    for (int ks = 0; ks < DF / 32; ++ks) {
      const int k0 = ks * 32;
      v16bf a = load_a_frag(arow, k0, hi);
#pragma unroll
      for (int ct = 0; ct < DB / 16; ++ct) {
        v16bf b = load_b_frag(Wb + (ct * 16 + ln) * SW + k0 + hi * 8);
        acc[ct] = __builtin_amdgcn_wmma_f32_16x16x32_bf16(false, a, false, b,
                                                          (short)0, acc[ct], false, false);
      }
    }
    // per-row gate/slot info for this tile's 8 rows of this lane half
    int   gv[8]; float gatev[8]; bool valid[8];
#pragma unroll
    for (int v = 0; v < 8; ++v) {
      int r = r0 + v + hi * 8;
      valid[v] = (r < m);
      int g = grouped[base + (valid[v] ? r : 0)];
      gv[v] = g;
      gatev[v] = top_w[(g >> 2) * TK + (g & 3)];
    }
#pragma unroll
    for (int ct = 0; ct < DB / 16; ++ct) {
      const int col = d0 + ct * 16 + ln;
      const float bov = bias_out[e * DM + col];
#pragma unroll
      for (int v = 0; v < 8; ++v) {
        if (valid[v]) {
          int g = gv[v];
          float o = (acc[ct][v] + bov) * gatev[v];
          contrib[((size_t)(g >> 2) * TK + (g & 3)) * DM + col] = o;
        }
      }
    }
  }
}

// ---- final combine: fixed-order sum over 4 slots -> deterministic -----------
__global__ void hexmoe_reduce(const float* __restrict__ contrib,
                              float* __restrict__ out, int total) {
  int i = blockIdx.x * blockDim.x + threadIdx.x;
  if (i >= total) return;
  int t = i / DM, d = i - t * DM;
  const float* c = contrib + (size_t)t * TK * DM + d;
  out[i] = ((c[0] + c[DM]) + c[2 * DM]) + c[3 * DM];
}

// ---- host launch ------------------------------------------------------------
extern "C" void kernel_launch(void* const* d_in, const int* in_sizes, int n_in,
                              void* d_out, int out_size, void* d_ws, size_t ws_size,
                              hipStream_t stream) {
  const float* x  = (const float*)d_in[0];
  const float* hw = (const float*)d_in[1];
  const float* Wi = (const float*)d_in[2];
  const float* bi = (const float*)d_in[3];
  const float* Wo = (const float*)d_in[4];
  const float* bo = (const float*)d_in[5];
  float* out = (float*)d_out;
  const int N = in_sizes[0] / DM;                 // B*T tokens

  char* ws = (char*)d_ws;
  float*  contrib = (float*)ws;  ws += (size_t)N * TK * DM * sizeof(float);
  __bf16* Xbf     = (__bf16*)ws; ws += (size_t)N * DM * sizeof(__bf16);
  __bf16* Hbuf    = (__bf16*)ws; ws += ((size_t)N * TK + 16) * DF * sizeof(__bf16);
  int*    counts  = (int*)ws;    ws += NE * sizeof(int);
  int*    cursors = (int*)ws;    ws += NE * sizeof(int);
  int*    offsets = (int*)ws;    ws += 80 * sizeof(int);
  int*    top_idx = (int*)ws;    ws += (size_t)N * TK * sizeof(int);
  float*  top_w   = (float*)ws;  ws += (size_t)N * TK * sizeof(float);
  int*    grouped = (int*)ws;

  hexmoe_zero<<<1, NE, 0, stream>>>(counts, cursors);
  hexmoe_route<<<(N + 255) / 256, 256, 0, stream>>>(hw, top_idx, top_w, counts, N);
  hexmoe_scan<<<1, 1, 0, stream>>>(counts, offsets);
  hexmoe_scatter<<<(N + 255) / 256, 256, 0, stream>>>(top_idx, offsets, cursors, grouped, N);
  hexmoe_xcast<<<(N * DM / 4 + 255) / 256, 256, 0, stream>>>(x, Xbf, N * DM);

  dim3 g1(NE, DF / FFB);   // 64 x 16
  moe_gemm1<<<g1, 256, LDS1_B, stream>>>(Xbf, Wi, bi, counts, offsets, grouped, Hbuf);

  dim3 g2(NE, DM / DB);    // 64 x 8
  moe_gemm2<<<g2, 256, LDS2_B, stream>>>(Hbuf, Wo, bo, counts, offsets, grouped,
                                         top_w, contrib);

  hexmoe_reduce<<<(N * DM + 255) / 256, 256, 0, stream>>>(contrib, out, N * DM);
}
